// AttentionBlock_14989435863732
// MI455X (gfx1250) — matrix operand order repaired and checked by reading, not compiled, not run on hardware
//
#include <hip/hip_runtime.h>
#include <hip/hip_bf16.h>
#include <math.h>

typedef __attribute__((ext_vector_type(16))) _Float16 v16h;
typedef __attribute__((ext_vector_type(8)))  float    v8f;

#define BB 4
#define LL 4096
#define DD 1024
#define HH 16
#define DKK 64
#define RR 2
#define BKT 64
#define NB 64      // LL / BKT
#define NPROJ 32   // NB / 2

union F4H8 { float4 f4; _Float16 h[8]; };

// ---------------------------------------------------------------------------
// WMMA fragment loaders, in contiguous k order (k = 16*(lane/16) + s), not the ISA table order
// A: 16x32 f16. lanes 0-15 = rows 0-15 with K {0..15}; lanes 16-31: +16.
// ---------------------------------------------------------------------------
static __device__ __forceinline__ v16h load_a_frag(const _Float16* A, int lda,
                                                   int row0, int k0, int lane) {
  const _Float16* p = A + (size_t)(row0 + (lane & 15)) * lda;
  int koff = (lane >> 4) << 4;      // 0 or 16
  v16h a;
#pragma unroll
  for (int g = 0; g < 4; ++g) {
    int k = k0 + koff + 2 * g;
    a[2 * g] = p[k]; a[2 * g + 1] = p[k + 1];
  }
#pragma unroll
  for (int g = 4; g < 8; ++g) {
    int k = k0 + koff + 2 * g;
    a[2 * g] = p[k]; a[2 * g + 1] = p[k + 1];
  }
  return a;
}

// B fragment (32x16, B[k][n]) sourced from a row-major S with B = S^T:
// B[k][n] = S[n][k]. Per lane this reads 16 CONTIGUOUS halves -> b128 loads.
static __device__ __forceinline__ v16h load_bT_frag(const _Float16* S, int lds,
                                                    int k0, int col0, int lane) {
  const _Float16* p = S + (size_t)(col0 + (lane & 15)) * lds;
  int kh = (lane >> 4) << 4;        // 0 or 16
  v16h b;
#pragma unroll
  for (int g = 0; g < 8; ++g) {
    int k = k0 + kh + 2 * g;
    b[2 * g] = p[k]; b[2 * g + 1] = p[k + 1];
  }
  return b;
}

// ---------------------------------------------------------------------------
// 1) LayerNorm fp32 -> fp16
// ---------------------------------------------------------------------------
__global__ __launch_bounds__(256)
void layernorm_kernel(const float* __restrict__ x, const float* __restrict__ gamma,
                      const float* __restrict__ beta, _Float16* __restrict__ xn) {
  __shared__ float s1[256], s2[256];
  int row = blockIdx.x;
  const float* xr = x + (size_t)row * DD;
  float vals[4], sum = 0.f, sq = 0.f;
#pragma unroll
  for (int i = 0; i < 4; ++i) {
    float v = xr[threadIdx.x + 256 * i];
    vals[i] = v; sum += v; sq += v * v;
  }
  s1[threadIdx.x] = sum; s2[threadIdx.x] = sq;
  __syncthreads();
  for (int s = 128; s > 0; s >>= 1) {
    if ((int)threadIdx.x < s) {
      s1[threadIdx.x] += s1[threadIdx.x + s];
      s2[threadIdx.x] += s2[threadIdx.x + s];
    }
    __syncthreads();
  }
  float mu = s1[0] * (1.0f / DD);
  float var = s2[0] * (1.0f / DD) - mu * mu;
  float rstd = rsqrtf(var + 1e-5f);
  _Float16* xo = xn + (size_t)row * DD;
#pragma unroll
  for (int i = 0; i < 4; ++i) {
    int c = threadIdx.x + 256 * i;
    xo[c] = (_Float16)((vals[i] - mu) * rstd * gamma[c] + beta[c]);
  }
}

// ---------------------------------------------------------------------------
// 2) Weight transpose + convert: Wt[n][k] = (f16)W[k][n]   (1024x1024)
// ---------------------------------------------------------------------------
__global__ __launch_bounds__(1024)
void transpose_w(const float* __restrict__ src, _Float16* __restrict__ dst) {
  __shared__ _Float16 tile[32][33];
  int x = blockIdx.x * 32 + threadIdx.x;   // n (fast in src)
  int y = blockIdx.y * 32 + threadIdx.y;   // k
  tile[threadIdx.y][threadIdx.x] = (_Float16)src[(size_t)y * DD + x];
  __syncthreads();
  int nx = blockIdx.y * 32 + threadIdx.x;  // k (fast in dst)
  int ny = blockIdx.x * 32 + threadIdx.y;  // n
  dst[(size_t)ny * DD + nx] = tile[threadIdx.x][threadIdx.y];
}

// ---------------------------------------------------------------------------
// 3) GEMM  C(MxN) = A(f16, MxK) * Wt^T(f16, Wt is NxK) + bias.
// mode 0: store fp16 into (B,H,L,DK) head layout.  mode 1: fp32 row-major.
// Block = 256 threads = 8 waves; each wave -> 32(M) x 64(N) tile.
// ---------------------------------------------------------------------------
__global__ __launch_bounds__(256)
void gemm_f16(const _Float16* __restrict__ A, const _Float16* __restrict__ Wt,
              const float* __restrict__ bias, int M, int K, int N,
              _Float16* __restrict__ out_h, float* __restrict__ out_f, int mode) {
  int lane = threadIdx.x & 31;
  int wave = threadIdx.x >> 5;
  int row0 = (blockIdx.x * 8 + wave) * 32;
  int col_base = blockIdx.y * 64;
  v8f acc[2][4] = {};
  const _Float16* pa = A + (size_t)(row0 + (lane & 15)) * K;
  for (int k0 = 0; k0 < K; k0 += 32) {
    if (k0 + 64 < K) __builtin_prefetch(pa + k0 + 64, 0, 1);   // global_prefetch_b8
    v16h a0 = load_a_frag(A, K, row0,      k0, lane);
    v16h a1 = load_a_frag(A, K, row0 + 16, k0, lane);
#pragma unroll
    for (int t = 0; t < 4; ++t) {
      v16h b = load_bT_frag(Wt, K, k0, col_base + t * 16, lane);
      acc[0][t] = __builtin_amdgcn_wmma_f32_16x16x32_f16(false, a0, false, b,
                                                         (short)0, acc[0][t], false, false);
      acc[1][t] = __builtin_amdgcn_wmma_f32_16x16x32_f16(false, a1, false, b,
                                                         (short)0, acc[1][t], false, false);
    }
  }
  int mr = (lane >> 4) << 3;
  int n_ = lane & 15;
#pragma unroll
  for (int s2 = 0; s2 < 2; ++s2) {
#pragma unroll
    for (int t = 0; t < 4; ++t) {
#pragma unroll
      for (int g = 0; g < 8; ++g) {
        int row = row0 + s2 * 16 + g + mr;
        int col = col_base + t * 16 + n_;
        float val = acc[s2][t][g] + bias[col];
        if (mode == 0) {
          int bi = row >> 12, l = row & (LL - 1);
          int h = col >> 6, dk = col & (DKK - 1);
          out_h[(((size_t)(bi * HH + h)) * LL + l) * DKK + dk] = (_Float16)val;
        } else {
          out_f[(size_t)row * N + col] = val;
        }
      }
    }
  }
}

// ---------------------------------------------------------------------------
// 4) LSH projection + argmax bucket -> sort keys (bucket*L + pos)
// rotations read contiguously & wave-uniform (all lanes share (r,h)).
// ---------------------------------------------------------------------------
__global__ __launch_bounds__(256)
void bucket_kernel(const _Float16* __restrict__ q, const float* __restrict__ rot,
                   int* __restrict__ keys) {
  int idx = blockIdx.x * 256 + threadIdx.x;        // (b, r, h, l)
  int l = idx & (LL - 1);
  int h = (idx >> 12) & (HH - 1);
  int r = (idx >> 16) & (RR - 1);
  int b = idx >> 17;
  const _Float16* qp = q + (((size_t)(b * HH + h)) * LL + l) * DKK;
  F4H8 uq[8];
  const float4* q4 = (const float4*)qp;
#pragma unroll
  for (int i = 0; i < 8; ++i) uq[i].f4 = q4[i];
  const float* rp = rot + ((size_t)(r * HH + h) * DKK) * NPROJ;
  float pj[NPROJ];
#pragma unroll
  for (int n = 0; n < NPROJ; ++n) pj[n] = 0.f;
  for (int d = 0; d < DKK; ++d) {
    float qd = (float)uq[d >> 3].h[d & 7];
    const float* rrow = rp + d * NPROJ;
#pragma unroll
    for (int n = 0; n < NPROJ; ++n) pj[n] += qd * rrow[n];
  }
  int best = 0; float bv = pj[0];
#pragma unroll
  for (int n = 1; n < NPROJ; ++n) if (pj[n] > bv) { bv = pj[n]; best = n; }
#pragma unroll
  for (int n = 0; n < NPROJ; ++n) if (-pj[n] > bv) { bv = -pj[n]; best = n + NPROJ; }
  keys[idx] = best * LL + l;
}

// ---------------------------------------------------------------------------
// 5) Per-(b,r,h) bitonic sort of 4096 (key, idx) in LDS; keys overwritten by order
// ---------------------------------------------------------------------------
__global__ __launch_bounds__(1024)
void sort_kernel(int* __restrict__ keys_order) {
  __shared__ int sk[LL];
  __shared__ int sv[LL];
  int* base = keys_order + (size_t)blockIdx.x * LL;
  for (int i = threadIdx.x; i < LL; i += 1024) { sk[i] = base[i]; sv[i] = i; }
  __syncthreads();
  for (int k = 2; k <= LL; k <<= 1) {
    for (int j = k >> 1; j > 0; j >>= 1) {
      for (int t = threadIdx.x; t < LL / 2; t += 1024) {
        int i1 = ((t & ~(j - 1)) << 1) | (t & (j - 1));
        int i2 = i1 | j;
        bool up = ((i1 & k) == 0);
        int k1 = sk[i1], k2 = sk[i2];
        if ((k1 > k2) == up) {
          sk[i1] = k2; sk[i2] = k1;
          int v1 = sv[i1]; sv[i1] = sv[i2]; sv[i2] = v1;
        }
      }
      __syncthreads();
    }
  }
  for (int i = threadIdx.x; i < LL; i += 1024) base[i] = sv[i];
}

// ---------------------------------------------------------------------------
// 6) Chunked LSH attention. One block (128 thr = 4 waves) per (b,r,h,chunk).
//    scores(64x128) = Q * K^T via WMMA; mask; softmax; P*V via WMMA;
//    scatter to original positions (fused "undo"), scatter LSE.
//    V is stored transposed in LDS so P*V B-fragments read contiguous runs.
// ---------------------------------------------------------------------------
__global__ __launch_bounds__(128)
void attn_kernel(const _Float16* __restrict__ q, const _Float16* __restrict__ v,
                 const int* __restrict__ order, const unsigned char* __restrict__ mask,
                 _Float16* __restrict__ oc, float* __restrict__ lse_ws) {
  __shared__ int      s_ord[128];
  __shared__ _Float16 s_q[64][80];      // 16B-aligned rows
  __shared__ _Float16 s_k[128][80];
  __shared__ _Float16 s_vt[DKK][136];   // V^T: [d][j]
  __shared__ _Float16 s_p[64][136];

  int c = blockIdx.x & (NB - 1);
  int h = (blockIdx.x >> 6) & (HH - 1);
  int r = (blockIdx.x >> 10) & (RR - 1);
  int b = blockIdx.x >> 11;
  int t = threadIdx.x;
  int lane = t & 31, wv = t >> 5;

  const int* ord = order + ((size_t)((b * RR + r) * HH + h)) * LL;
  int cp = (c + NB - 1) & (NB - 1);                 // lookback chunk (roll by 1)
  int src = (t < 64) ? (cp * BKT + t) : (c * BKT + (t - 64));
  s_ord[t] = ord[src];

  // gather one kk/vv row per thread; rows 64..127 are the current (query) chunk
  int p = s_ord[t];
  const float4* q4 = (const float4*)(q + (((size_t)(b * HH + h)) * LL + p) * DKK);
  const float4* v4 = (const float4*)(v + (((size_t)(b * HH + h)) * LL + p) * DKK);
  F4H8 uq[8], uv[8];
#pragma unroll
  for (int i = 0; i < 8; ++i) { uq[i].f4 = q4[i]; uv[i].f4 = v4[i]; }
  float ss = 0.f;
#pragma unroll
  for (int i = 0; i < 8; ++i)
#pragma unroll
    for (int j = 0; j < 8; ++j) { float qv = (float)uq[i].h[j]; ss += qv * qv; }
  float inv = 1.0f / (sqrtf(ss) + 1e-9f);
#pragma unroll
  for (int i = 0; i < 8; ++i) {
    F4H8 kk;
#pragma unroll
    for (int j = 0; j < 8; ++j) {
      kk.h[j] = (_Float16)((float)uq[i].h[j] * inv);
      s_vt[i * 8 + j][t] = uv[i].h[j];
    }
    ((float4*)&s_k[t][0])[i] = kk.f4;               // ds_store_b128
    if (t >= 64) ((float4*)&s_q[t - 64][0])[i] = uq[i].f4;
  }
  __syncthreads();

  // scores: each wave does 16 query rows x 128 key cols
  int row0 = wv * 16;
  v8f sc[8] = {};
  for (int k0 = 0; k0 < DKK; k0 += 32) {
    v16h a = load_a_frag(&s_q[0][0], 80, row0, k0, lane);
#pragma unroll
    for (int tn = 0; tn < 8; ++tn) {
      v16h bf = load_bT_frag(&s_k[0][0], 80, k0, tn * 16, lane);
      sc[tn] = __builtin_amdgcn_wmma_f32_16x16x32_f16(false, a, false, bf,
                                                      (short)0, sc[tn], false, false);
    }
  }

  const unsigned char* mrow = mask + (size_t)b * LL;
  int mr = (lane >> 4) << 3;
  int n_ = lane & 15;
  const float scale = 0.125f;                       // 1/sqrt(64)
#pragma unroll
  for (int tn = 0; tn < 8; ++tn) {
    int kj = s_ord[tn * 16 + n_];
    bool valid = mrow[kj] != 0;
#pragma unroll
    for (int g = 0; g < 8; ++g) {
      int qi = s_ord[64 + row0 + g + mr];
      float sv2 = sc[tn][g] * scale;
      if (!((qi >= kj) && valid)) sv2 = -1e9f;      // causal + mask
      if (qi == kj) sv2 = -1e5f;                    // exclude self
      sc[tn][g] = sv2;
    }
  }

  // per-row softmax: row spans 16 lanes x 8 frags; xor-shfl within 16-lane half
  float* lse_out = lse_ws + ((size_t)((b * RR + r) * HH + h)) * LL;
#pragma unroll
  for (int g = 0; g < 8; ++g) {
    float mx = -3.0e38f;
#pragma unroll
    for (int tn = 0; tn < 8; ++tn) mx = fmaxf(mx, sc[tn][g]);
    for (int off = 1; off < 16; off <<= 1) mx = fmaxf(mx, __shfl_xor(mx, off, 32));
    float sum = 0.f;
#pragma unroll
    for (int tn = 0; tn < 8; ++tn) sum += __expf(sc[tn][g] - mx);
    for (int off = 1; off < 16; off <<= 1) sum += __shfl_xor(sum, off, 32);
    float invs = 1.0f / sum;
    int i = row0 + g + mr;
#pragma unroll
    for (int tn = 0; tn < 8; ++tn)
      s_p[i][tn * 16 + n_] = (_Float16)(__expf(sc[tn][g] - mx) * invs);
    if (n_ == 0) lse_out[s_ord[64 + i]] = mx + __logf(sum);
  }
  __syncthreads();

  // out = P(64x128) * V(128x64);  B[k=j][n=d] = s_vt[d][j] -> contiguous k
  v8f oacc[4] = {};
  for (int k0 = 0; k0 < 128; k0 += 32) {
    v16h a = load_a_frag(&s_p[0][0], 136, row0, k0, lane);
#pragma unroll
    for (int tn = 0; tn < 4; ++tn) {
      v16h bf = load_bT_frag(&s_vt[0][0], 136, k0, tn * 16, lane);
      oacc[tn] = __builtin_amdgcn_wmma_f32_16x16x32_f16(false, a, false, bf,
                                                        (short)0, oacc[tn], false, false);
    }
  }

  _Float16* ocp = oc + ((size_t)((b * RR + r) * HH + h)) * LL * DKK;
#pragma unroll
  for (int g = 0; g < 8; ++g) {
    int l = s_ord[64 + row0 + g + mr];              // fused undo-scatter
#pragma unroll
    for (int tn = 0; tn < 4; ++tn)
      ocp[(size_t)l * DKK + tn * 16 + n_] = (_Float16)oacc[tn][g];
  }
}

// ---------------------------------------------------------------------------
// 7) Combine rounds with softmax(LSE) weights -> fp16 (B*L, D) activation
// ---------------------------------------------------------------------------
__global__ __launch_bounds__(256)
void combine_kernel(const _Float16* __restrict__ oc, const float* __restrict__ lse,
                    _Float16* __restrict__ xo) {
  int idx = blockIdx.x * 256 + threadIdx.x;        // (b, h, l, d)
  int d = idx & (DKK - 1);
  int l = (idx >> 6) & (LL - 1);
  int h = (idx >> 18) & (HH - 1);
  int b = idx >> 22;
  size_t base0 = ((size_t)((b * RR + 0) * HH + h)) * LL + l;
  size_t base1 = base0 + (size_t)HH * LL;
  float l0 = lse[base0], l1 = lse[base1];
  float m = fmaxf(l0, l1);
  float w0 = __expf(l0 - m), w1 = __expf(l1 - m);
  float invs = 1.0f / (w0 + w1);
  float o0 = (float)oc[base0 * DKK + d];
  float o1 = (float)oc[base1 * DKK + d];
  xo[((size_t)(b * LL + l)) * DD + h * DKK + d] = (_Float16)((o0 * w0 + o1 * w1) * invs);
}

// ---------------------------------------------------------------------------
extern "C" void kernel_launch(void* const* d_in, const int* in_sizes, int n_in,
                              void* d_out, int out_size, void* d_ws, size_t ws_size,
                              hipStream_t stream) {
  const float* x     = (const float*)d_in[0];
  const float* gamma = (const float*)d_in[1];
  const float* beta  = (const float*)d_in[2];
  const float* Wq    = (const float*)d_in[3];
  const float* bq    = (const float*)d_in[4];
  const float* Wv    = (const float*)d_in[5];
  const float* bv    = (const float*)d_in[6];
  const float* Wo    = (const float*)d_in[7];
  const float* bo    = (const float*)d_in[8];
  const float* rot   = (const float*)d_in[9];
  const unsigned char* mask = (const unsigned char*)d_in[10];
  float* out = (float*)d_out;

  char* ws = (char*)d_ws;
  size_t off = 0;
  auto alloc = [&](size_t bytes) {
    char* pp = ws + off;
    off += (bytes + 255) & ~(size_t)255;
    return pp;
  };
  _Float16* xn  = (_Float16*)alloc((size_t)BB * LL * DD * 2);       // reused as xo later
  _Float16* qh  = (_Float16*)alloc((size_t)BB * LL * DD * 2);
  _Float16* vh  = (_Float16*)alloc((size_t)BB * LL * DD * 2);
  _Float16* wqt = (_Float16*)alloc((size_t)DD * DD * 2);
  _Float16* wvt = (_Float16*)alloc((size_t)DD * DD * 2);
  _Float16* wot = (_Float16*)alloc((size_t)DD * DD * 2);
  int*      ordb = (int*)alloc((size_t)BB * RR * HH * LL * 4);
  float*    lse  = (float*)alloc((size_t)BB * RR * HH * LL * 4);
  _Float16* oc   = (_Float16*)alloc((size_t)BB * RR * HH * LL * DKK * 2);

  layernorm_kernel<<<BB * LL, 256, 0, stream>>>(x, gamma, beta, xn);
  dim3 tgrid(DD / 32, DD / 32), tblk(32, 32);
  transpose_w<<<tgrid, tblk, 0, stream>>>(Wq, wqt);
  transpose_w<<<tgrid, tblk, 0, stream>>>(Wv, wvt);
  transpose_w<<<tgrid, tblk, 0, stream>>>(Wo, wot);

  dim3 ggrid(BB * LL / 256, DD / 64);
  gemm_f16<<<ggrid, 256, 0, stream>>>(xn, wqt, bq, BB * LL, DD, DD, qh, nullptr, 0);
  gemm_f16<<<ggrid, 256, 0, stream>>>(xn, wvt, bv, BB * LL, DD, DD, vh, nullptr, 0);

  bucket_kernel<<<(BB * RR * HH * LL) / 256, 256, 0, stream>>>(qh, rot, ordb);
  sort_kernel<<<BB * RR * HH, 1024, 0, stream>>>(ordb);
  attn_kernel<<<BB * RR * HH * NB, 128, 0, stream>>>(qh, vh, ordb, mask, oc, lse);
  combine_kernel<<<(BB * HH * LL * DKK) / 256, 256, 0, stream>>>(oc, lse, xn);

  gemm_f16<<<ggrid, 256, 0, stream>>>(xn, wot, bo, BB * LL, DD, DD, nullptr, out, 1);
}
